// AHA_linearDG_kSparseInhibition_87144886436101
// MI455X (gfx1250) — matrix operand order repaired and checked by reading, not compiled, not run on hardware
//
#include <hip/hip_runtime.h>
#include <stdint.h>

// ---------------------------------------------------------------------------
// encoding = inputs[256x2048] @ weight[16384x2048]^T (fp32), then sequential
// per-row top-50 + inhibition scan.
//
// MI455X: GEMM reads fp32 once from HBM (146MB -> ~6.3us @23.3TB/s), converts
// to bf16 via v_cvt_pk_bf16_f32, stages in double-buffered LDS, computes with
// v_wmma_f32_16x16x32_bf16 (fp32 accum). Wave tile 32x64 = 8 WMMA/K-step with
// B-fragment reuse; one barrier per K-step. Weight loads are non-temporal
// (streamed once); inputs stay hot in the 192MB L2.
// Scan: serial by construction -> one 1024-thread WG, registers + radix select.
// ---------------------------------------------------------------------------

typedef __attribute__((ext_vector_type(16))) __bf16 v16bf;
typedef __attribute__((ext_vector_type(2)))  __bf16 v2bf;
typedef __attribute__((ext_vector_type(8)))  float  v8f;
typedef __attribute__((ext_vector_type(4)))  float  v4f;   // native vec for NT loads

#define M_DIM 256
#define K_DIM 2048
#define N_DIM 16384

#define BM 64
#define BN 256
#define BK 32
#define KSTEPS (K_DIM / BK)

#define LDA  20            // uints per A row: 16 (=32 bf16) + 4 pad
#define LDBU (BN / 2 + 4)  // uints per B k-row: 128 N-pairs + 4 pad

#define TOPK 50
#define DECAY 0.95f
#define SCAN_T 1024
#define EPT (N_DIM / SCAN_T)

__device__ __forceinline__ uint32_t pk_bf16(float lo, float hi) {
#if defined(__has_builtin)
#if __has_builtin(__builtin_amdgcn_cvt_pk_bf16_f32)
  auto r = __builtin_amdgcn_cvt_pk_bf16_f32(lo, hi);
  static_assert(sizeof(r) == 4, "cvt_pk_bf16_f32 must return 32 bits");
  union { decltype(r) v; uint32_t u; } c;
  c.v = r;
  return c.u;
#else
  // build-vector of two casts: canonical pattern for v_cvt_pk_bf16_f32
  v2bf v;
  v.x = (__bf16)lo;
  v.y = (__bf16)hi;
  union { v2bf v; uint32_t u; } c;
  c.v = v;
  return c.u;
#endif
#else
  v2bf v;
  v.x = (__bf16)lo;
  v.y = (__bf16)hi;
  union { v2bf v; uint32_t u; } c;
  c.v = v;
  return c.u;
#endif
}

__device__ __forceinline__ uint32_t float_key(float x) {
  uint32_t u = __float_as_uint(x);
  return u ^ ((uint32_t)((int32_t)u >> 31) | 0x80000000u);
}

// ---------------------------------------------------------------------------
// GEMM: out[m][n] = sum_k inputs[m][k] * weight[n][k].
// 256 threads = 8 waves as 2(M)x4(N); wave computes 32x64 (8 WMMA / K-step).
// ---------------------------------------------------------------------------
__global__ __launch_bounds__(256)
void dg_gemm_bf16_wmma(const float* __restrict__ inp,
                       const float* __restrict__ wgt,
                       float* __restrict__ enc) {
  __shared__ uint32_t As[2][BM * LDA];   // m-major, bf16 pairs along K
  __shared__ uint32_t Bs[2][BK * LDBU];  // k-major, bf16 pairs along N

  const int tid  = threadIdx.x;
  const int lane = tid & 31;
  const int wid  = tid >> 5;
  const int wm   = wid & 1;    // M group (32 rows)
  const int wn   = wid >> 1;   // N group (64 cols)
  const int m0   = blockIdx.y * BM;
  const int n0   = blockIdx.x * BN;

  // A fetch: 64x32 tile, 8 consecutive floats per thread.
  const int a_row  = tid >> 2;
  const int a_colf = (tid & 3) * 8;
  // B fetch: 256x32 tile, 4 adjacent rows x 8 consecutive k per thread
  // (adjacent-N pairs pack into uints -> wide ds stores).
  const int b_q    = tid >> 2;        // 0..63 -> rows 4q..4q+3
  const int b_colf = (tid & 3) * 8;   // k chunk

  const float* Ap = inp + (size_t)(m0 + a_row) * K_DIM + a_colf;
  const float* Bp = wgt + (size_t)(n0 + 4 * b_q) * K_DIM + b_colf;

  float a_reg[8];
  float b_reg[32];  // [row 0..3][k 0..7]

  auto fetch = [&](int k0) {
    const v4f* ap = (const v4f*)(Ap + k0);
    v4f a0 = ap[0], a1 = ap[1];
    a_reg[0]=a0.x; a_reg[1]=a0.y; a_reg[2]=a0.z; a_reg[3]=a0.w;
    a_reg[4]=a1.x; a_reg[5]=a1.y; a_reg[6]=a1.z; a_reg[7]=a1.w;
#pragma unroll
    for (int r = 0; r < 4; ++r) {
      const v4f* bp = (const v4f*)(Bp + (size_t)r * K_DIM + k0);
      v4f v0 = __builtin_nontemporal_load(bp);      // weight: streamed once
      v4f v1 = __builtin_nontemporal_load(bp + 1);
      b_reg[r*8+0]=v0.x; b_reg[r*8+1]=v0.y; b_reg[r*8+2]=v0.z; b_reg[r*8+3]=v0.w;
      b_reg[r*8+4]=v1.x; b_reg[r*8+5]=v1.y; b_reg[r*8+6]=v1.z; b_reg[r*8+7]=v1.w;
    }
  };

  auto stage = [&](int buf) {
    uint32_t* adst = &As[buf][a_row * LDA + (a_colf >> 1)];
#pragma unroll
    for (int j = 0; j < 4; ++j)
      adst[j] = pk_bf16(a_reg[2*j], a_reg[2*j+1]);
#pragma unroll
    for (int k = 0; k < 8; ++k) {
      uint2 v;
      v.x = pk_bf16(b_reg[0*8+k], b_reg[1*8+k]);   // N pair (4q, 4q+1)
      v.y = pk_bf16(b_reg[2*8+k], b_reg[3*8+k]);   // N pair (4q+2, 4q+3)
      *(uint2*)&Bs[buf][(b_colf + k) * LDBU + 2 * b_q] = v;
    }
  };

  union Frag { uint32_t u[8]; v16bf v; };

  v8f acc[2][4] = {};

  const int half = lane >> 4;
  const int lrow = lane & 15;

  fetch(0);
  stage(0);
  __syncthreads();

  for (int t = 0; t < KSTEPS; ++t) {
    const int cur = t & 1;
    const bool more = (t + 1 < KSTEPS);
    if (more) fetch((t + 1) * BK);   // global loads overlap compute below

    Frag fa[2];
#pragma unroll
    for (int g = 0; g < 2; ++g) {
      const uint32_t* aB = &As[cur][(wm * 32 + g * 16 + lrow) * LDA];
#pragma unroll
      for (int j = 0; j < 8; ++j)    // K pairs: lanes<16 {0..7,16..23}, else {8..15,24..31}
        fa[g].u[j] = aB[((j >> 2) * 8) + half * 4 + (j & 3)];
    }
#pragma unroll
    for (int s = 0; s < 4; ++s) {
      Frag fb;
      const int nn = wn * 64 + s * 16 + lrow;
      const uint32_t* bp = &Bs[cur][nn >> 1];
      const int sh = (nn & 1) * 16;
#pragma unroll
      for (int j = 0; j < 8; ++j) {
        const int kk = (j >> 2) * 16 + half * 8 + (j & 3) * 2;
        fb.u[j] = ((bp[kk * LDBU] >> sh) & 0xFFFFu) | ((bp[(kk + 1) * LDBU] >> sh) << 16);
      }
      acc[0][s] = __builtin_amdgcn_wmma_f32_16x16x32_bf16(
          false, fa[0].v, false, fb.v, (short)0, acc[0][s], false, false);
      acc[1][s] = __builtin_amdgcn_wmma_f32_16x16x32_bf16(
          false, fa[1].v, false, fb.v, (short)0, acc[1][s], false, false);
    }

    if (more) stage(cur ^ 1);        // stage next tile into other buffer
    __syncthreads();
  }

  // C layout: VGPR j -> M = j + 8*(lane>>4), N = lane&15.
  const int orow0 = m0 + wm * 32 + half * 8;
  const int ocol0 = n0 + wn * 64 + lrow;
#pragma unroll
  for (int g = 0; g < 2; ++g)
#pragma unroll
    for (int s = 0; s < 4; ++s)
#pragma unroll
      for (int j = 0; j < 8; ++j)
        enc[(size_t)(orow0 + g * 16 + j) * N_DIM + ocol0 + s * 16] = acc[g][s][j];
}

// ---------------------------------------------------------------------------
// Sequential top-K + inhibition scan. One WG, 1024 threads, in-place on enc.
// ---------------------------------------------------------------------------
__global__ __launch_bounds__(SCAN_T)
void dg_sparse_inhibit(float* __restrict__ enc) {
  __shared__ int hist1[256];
  __shared__ int hist2[256];
  __shared__ int s_wsum[32];
  __shared__ int s_bin1, s_g1, s_th16, s_R;

  const int tid  = threadIdx.x;
  const int lane = tid & 31;
  const int wid  = tid >> 5;
  const int cbase = tid * EPT;

  float inh[EPT];
#pragma unroll
  for (int i = 0; i < EPT; ++i) inh[i] = 0.0f;

  for (int b = 0; b < M_DIM; ++b) {
    float* row = enc + (size_t)b * N_DIM;

    float e[EPT];
    uint32_t key[EPT];
    {
      const v4f* rp = (const v4f*)(row + cbase);
#pragma unroll
      for (int j = 0; j < 4; ++j) {
        v4f v = rp[j];
        e[4*j+0]=v.x; e[4*j+1]=v.y; e[4*j+2]=v.z; e[4*j+3]=v.w;
      }
    }
#pragma unroll
    for (int i = 0; i < EPT; ++i)
      key[i] = float_key(fabsf(e[i]) * (1.0f - inh[i]));

    if (tid < 256) { hist1[tid] = 0; hist2[tid] = 0; }
    __syncthreads();

#pragma unroll
    for (int i = 0; i < EPT; ++i) atomicAdd(&hist1[key[i] >> 24], 1);
    __syncthreads();

    if (wid == 0) {  // pass 1: top-8-bit threshold bin (wave0, shfl suffix scan)
      int h[8], lsum = 0;
#pragma unroll
      for (int j = 0; j < 8; ++j) { h[j] = hist1[lane * 8 + j]; lsum += h[j]; }
      int incl = lsum;
      for (int off = 1; off < 32; off <<= 1) {
        int n = __shfl_down(incl, off, 32);
        if (lane + off < 32) incl += n;
      }
      int cum = incl - lsum;
      for (int j = 7; j >= 0; --j) {
        int c = h[j];
        if (cum < TOPK && cum + c >= TOPK) { s_bin1 = lane * 8 + j; s_g1 = cum; }
        cum += c;
      }
    }
    __syncthreads();

    const uint32_t bin1 = (uint32_t)s_bin1;
#pragma unroll
    for (int i = 0; i < EPT; ++i)
      if ((key[i] >> 24) == bin1) atomicAdd(&hist2[(key[i] >> 16) & 0xFF], 1);
    __syncthreads();

    if (wid == 0) {  // pass 2: refine on next 8 bits
      int g1 = s_g1;
      int h[8], lsum = 0;
#pragma unroll
      for (int j = 0; j < 8; ++j) { h[j] = hist2[lane * 8 + j]; lsum += h[j]; }
      int incl = lsum;
      for (int off = 1; off < 32; off <<= 1) {
        int n = __shfl_down(incl, off, 32);
        if (lane + off < 32) incl += n;
      }
      int cum = (incl - lsum) + g1;
      for (int j = 7; j >= 0; --j) {
        int c = h[j];
        if (cum < TOPK && cum + c >= TOPK) {
          s_th16 = ((int)bin1 << 8) | (lane * 8 + j);
          s_R    = TOPK - cum;
        }
        cum += c;
      }
    }
    __syncthreads();

    const uint32_t th16 = (uint32_t)s_th16;
    const int R = s_R;

    // deterministic rank of tied elements: block exclusive scan (lowest index)
    int myeq = 0;
#pragma unroll
    for (int i = 0; i < EPT; ++i) myeq += ((key[i] >> 16) == th16);
    int incl = myeq;
    for (int off = 1; off < 32; off <<= 1) {
      int n = __shfl_up(incl, off, 32);
      if (lane >= off) incl += n;
    }
    const int wexcl = incl - myeq;
    if (lane == 31) s_wsum[wid] = incl;
    __syncthreads();
    if (wid == 0) {
      int v = s_wsum[lane];
      int inc2 = v;
      for (int off = 1; off < 32; off <<= 1) {
        int n = __shfl_up(inc2, off, 32);
        if (lane >= off) inc2 += n;
      }
      s_wsum[lane] = inc2 - v;
    }
    __syncthreads();

    int slot = s_wsum[wid] + wexcl;
    float o[EPT];
#pragma unroll
    for (int i = 0; i < EPT; ++i) {
      uint32_t k16 = key[i] >> 16;
      bool fire;
      if (k16 > th16)       fire = true;
      else if (k16 == th16) { fire = (slot < R); ++slot; }
      else                  fire = false;
      o[i]   = fire ? e[i] : 0.0f;
      inh[i] = inh[i] * DECAY + (fire ? 1.0f : 0.0f);
    }
    {
      v4f* rp = (v4f*)(row + cbase);
#pragma unroll
      for (int j = 0; j < 4; ++j) {
        v4f v;
        v.x = o[4*j+0]; v.y = o[4*j+1]; v.z = o[4*j+2]; v.w = o[4*j+3];
        rp[j] = v;
      }
    }
    __syncthreads();
  }
}

extern "C" void kernel_launch(void* const* d_in, const int* in_sizes, int n_in,
                              void* d_out, int out_size, void* d_ws, size_t ws_size,
                              hipStream_t stream) {
  (void)in_sizes; (void)n_in; (void)out_size; (void)d_ws; (void)ws_size;
  const float* inputs = (const float*)d_in[0];   // [256, 2048]
  const float* weight = (const float*)d_in[1];   // [16384, 2048]
  float* out = (float*)d_out;                    // [256, 16384]

  dim3 grid(N_DIM / BN, M_DIM / BM);             // 64 x 4 workgroups
  dg_gemm_bf16_wmma<<<grid, 256, 0, stream>>>(inputs, weight, out);
  dg_sparse_inhibit<<<1, SCAN_T, 0, stream>>>(out);  // in-place filter
}
